// INGPNetworkRHINO_56547539419223
// MI455X (gfx1250) — matrix operand order repaired and checked by reading, not compiled, not run on hardware
//
#include <hip/hip_runtime.h>
#include <hip/hip_bf16.h>

// ---------------------------------------------------------------------------
// Fused Instant-NGP forward for MI455X (gfx1250, wave32, WMMA).
// One wave = one 16-point WMMA M-tile through the whole network.
// All GEMMs use v_wmma_f32_16x16x32_f16 with f16 weights pre-transposed in LDS.
// ---------------------------------------------------------------------------

typedef __attribute__((ext_vector_type(16))) _Float16 v16h;
typedef __attribute__((ext_vector_type(8)))  _Float16 v8h;
typedef __attribute__((ext_vector_type(8)))  float    v8f;

#define NPTS   1048576
#define TSIZE  (1u << 19)
#define SEXP   (7.0f / 15.0f)   // log2(2048/16)/(16-1)

// ---- LDS layout (offsets in _Float16 units) ----
#define OFF_TW0T 0                         // [64][64]   t-branch weights^T
#define OFF_W0T  (OFF_TW0T + 64 * 64)      // [128][96]
#define OFF_W1T  (OFF_W0T + 128 * 96)      // [128][128]
#define OFF_W2T  (OFF_W1T + 128 * 128)     // [128][224]
#define OFF_W3T  (OFF_W2T + 128 * 224)     // [128][224]
#define OFF_W4T  (OFF_W3T + 128 * 224)     // [16][128]  (padded 2->16 outs)
#define OFF_BIAS (OFF_W4T + 16 * 128)      // 592 floats: tb0|b0|b1|b2|b3|b4pad
#define BIAS_F32 592
#define OFF_ACT  (OFF_BIAS + BIAS_F32 * 2)
#define WAVE_ACT_HALFS (16 * 224 + 16 * 128)
#define SMEM_BYTES ((size_t)(OFF_ACT + 8 * WAVE_ACT_HALFS) * 2)  // 276,800 B < 320 KB/WGP

struct ASrc { const _Float16* base; int pitch; };

// A-tile (16x32 f16): lane&15 = M row; lane>=16 takes K halves {8..15,24..31}.
__device__ __forceinline__ v16h lds_loadA(const _Float16* base, int pitch, int lane) {
  const int m  = lane & 15;
  const int ks = (lane >> 4) << 3;                 // 0 or 8
  const _Float16* p = base + m * pitch + ks;
  union { v16h v; v8h h[2]; } a;
  a.h[0] = *(const v8h*)(p);                       // K = ks..ks+7
  a.h[1] = *(const v8h*)(p + 16);                  // K = 16+ks..16+ks+7
  return a.v;
}

// B-tile (32x16 f16) from weights stored transposed [out][in]:
// lane&15 = N col; lane>=16 takes K halves {8..15,24..31} (matches A-tile).
__device__ __forceinline__ v16h lds_loadB(const _Float16* wt, int pitch, int ntile,
                                          int kbase, int lane) {
  const int n  = lane & 15;
  const int kk = (lane >> 4) << 3;                 // 0 or 8
  const _Float16* p = wt + (ntile * 16 + n) * pitch + kbase + kk;
  union { v16h v; v8h h[2]; } b;
  b.h[0] = *(const v8h*)(p);
  b.h[1] = *(const v8h*)(p + 16);
  return b.v;
}

// One MLP layer: D = relu(A @ W^T + b), A from LDS chunks, out -> LDS f16.
template <int NCH, int NT>
__device__ __forceinline__ void do_layer(const ASrc (&asrc)[NCH],
                                         const _Float16* wt, int wpitch,
                                         const float* bias,
                                         _Float16* dst, int dpitch) {
  const int lane = threadIdx.x & 31;
  const int n    = lane & 15;
  const int mb   = (lane >> 4) << 3;
#pragma unroll
  for (int nt = 0; nt < NT; ++nt) {
    const float bv = bias[nt * 16 + n];
    v8f c;
#pragma unroll
    for (int r = 0; r < 8; ++r) c[r] = bv;
#pragma unroll
    for (int kc = 0; kc < NCH; ++kc) {
      v16h a = lds_loadA(asrc[kc].base, asrc[kc].pitch, lane);
      v16h b = lds_loadB(wt, wpitch, nt, kc * 32, lane);
      c = __builtin_amdgcn_wmma_f32_16x16x32_f16(false, a, false, b, (short)0, c,
                                                 false, false);
    }
#pragma unroll
    for (int r = 0; r < 8; ++r) {
      float v = fmaxf(c[r], 0.0f);
      dst[(mb + r) * dpitch + nt * 16 + n] = (_Float16)v;
    }
  }
}

// Transpose-convert fp32 [K][N] row-major -> f16 [Npad][Kpad] (WMMA B friendly).
__device__ __forceinline__ void stage_w(const float* __restrict__ src, _Float16* dst,
                                        int K, int N, int Kpad, int Npad, int tid) {
  for (int i = tid; i < Npad * Kpad; i += 256) {
    const int n = i / Kpad;
    const int k = i - n * Kpad;
    const float v = (k < K && n < N) ? src[k * N + n] : 0.0f;
    dst[i] = (_Float16)v;
  }
}

__global__ void __launch_bounds__(256, 1)
ingp_fused(const float* __restrict__ x,   const float* __restrict__ emb,
           const float* __restrict__ tw0, const float* __restrict__ tb0,
           const float* __restrict__ w0,  const float* __restrict__ b0,
           const float* __restrict__ w1,  const float* __restrict__ b1,
           const float* __restrict__ w2,  const float* __restrict__ b2,
           const float* __restrict__ w3,  const float* __restrict__ b3,
           const float* __restrict__ w4,  const float* __restrict__ b4,
           float* __restrict__ out) {
  extern __shared__ char smem_raw[];
  _Float16* smem = (_Float16*)smem_raw;
  _Float16* tw0t = smem + OFF_TW0T;
  _Float16* w0t  = smem + OFF_W0T;
  _Float16* w1t  = smem + OFF_W1T;
  _Float16* w2t  = smem + OFF_W2T;
  _Float16* w3t  = smem + OFF_W3T;
  _Float16* w4t  = smem + OFF_W4T;
  float*    bias = (float*)(smem + OFF_BIAS);

  const int tid = threadIdx.x;

  // ---- stage all weights once per block (f16, transposed) ----
  stage_w(tw0, tw0t, 57, 64, 64, 64, tid);
  stage_w(w0,  w0t,  96, 128, 96, 128, tid);
  stage_w(w1,  w1t, 128, 128, 128, 128, tid);
  stage_w(w2,  w2t, 224, 128, 224, 128, tid);
  stage_w(w3,  w3t, 224, 128, 224, 128, tid);
  stage_w(w4,  w4t, 128,   2, 128,  16, tid);
  for (int i = tid; i < 64;  i += 256) bias[i] = tb0[i];
  for (int i = tid; i < 128; i += 256) {
    bias[64  + i] = b0[i];
    bias[192 + i] = b1[i];
    bias[320 + i] = b2[i];
    bias[448 + i] = b3[i];
  }
  for (int i = tid; i < 16; i += 256) bias[576 + i] = (i < 2) ? b4[i] : 0.0f;
  __syncthreads();

  const int wave = tid >> 5;
  const int lane = tid & 31;
  const int p    = lane & 15;   // point within this wave's 16-point tile
  const int hi   = lane >> 4;   // 0/1: split of levels / freq features
  _Float16* act  = smem + OFF_ACT + wave * WAVE_ACT_HALFS;  // [16][224]: 0..127 cur, 128..223 h|t
  _Float16* act2 = act + 16 * 224;                          // [16][128]: freq feats / ping-pong

  const int TILES = NPTS / 128;
  for (int tile = blockIdx.x; tile < TILES; tile += gridDim.x) {
    const int pbase = tile * 128 + wave * 16;
    const int gp    = pbase + p;
    const float xx = x[3 * gp + 0], xy = x[3 * gp + 1], xz = x[3 * gp + 2];

    // ---- hash-grid encode: 8 levels per lane, trilinear over 8 corners ----
    const int l0 = hi * 8;
#pragma unroll
    for (int li = 0; li < 8; ++li) {
      const int l = l0 + li;
      const float scale = exp2f((float)l * SEXP) * 16.0f - 1.0f;
      const int res = (int)ceilf(scale) + 1;
      const bool linear = ((long long)res * res * res) <= (long long)TSIZE;
      const float px = xx * scale + 0.5f, py = xy * scale + 0.5f, pz = xz * scale + 0.5f;
      const float fx = floorf(px), fy = floorf(py), fz = floorf(pz);
      const float rx = px - fx, ry = py - fy, rz = pz - fz;
      const unsigned ix = (unsigned)fx, iy = (unsigned)fy, iz = (unsigned)fz;
      const float* tab = emb + (size_t)l * TSIZE * 2;
      float f0 = 0.0f, f1 = 0.0f;
#pragma unroll
      for (int c = 0; c < 8; ++c) {
        const unsigned cx = ix + (c & 1), cy = iy + ((c >> 1) & 1), cz = iz + ((c >> 2) & 1);
        unsigned idx;
        if (linear) idx = cx + cy * (unsigned)res + cz * (unsigned)(res * res);
        else        idx = cx ^ (cy * 2654435761u) ^ (cz * 805459861u);
        idx &= (TSIZE - 1u);
        const float wx = (c & 1)        ? rx : 1.0f - rx;
        const float wy = ((c >> 1) & 1) ? ry : 1.0f - ry;
        const float wz = ((c >> 2) & 1) ? rz : 1.0f - rz;
        const float w = wx * wy * wz;
        f0 += w * tab[2 * idx];
        f1 += w * tab[2 * idx + 1];
      }
      union { _Float16 h[2]; unsigned u; } pk;
      pk.h[0] = (_Float16)f0; pk.h[1] = (_Float16)f1;
      *(unsigned*)&act[p * 224 + 128 + 2 * l] = pk.u;   // h -> cols 128..159
    }

    // ---- freq encode -> act2[p][0..63] (57 real + 7 zero pad) ----
    {
      _Float16* fr = act2 + p * 128;
      if (hi == 0) {
        fr[0] = (_Float16)xx; fr[1] = (_Float16)xy; fr[2] = (_Float16)xz;
#pragma unroll
        for (int i = 0; i < 4; ++i) {
          const float m = (float)(1 << i);
          fr[3 + 6 * i + 0] = (_Float16)sinf(xx * m);
          fr[3 + 6 * i + 1] = (_Float16)sinf(xy * m);
          fr[3 + 6 * i + 2] = (_Float16)sinf(xz * m);
          fr[3 + 6 * i + 3] = (_Float16)cosf(xx * m);
          fr[3 + 6 * i + 4] = (_Float16)cosf(xy * m);
          fr[3 + 6 * i + 5] = (_Float16)cosf(xz * m);
        }
      } else {
#pragma unroll
        for (int i = 4; i < 9; ++i) {
          const float m = (float)(1 << i);
          fr[3 + 6 * i + 0] = (_Float16)sinf(xx * m);
          fr[3 + 6 * i + 1] = (_Float16)sinf(xy * m);
          fr[3 + 6 * i + 2] = (_Float16)sinf(xz * m);
          fr[3 + 6 * i + 3] = (_Float16)cosf(xx * m);
          fr[3 + 6 * i + 4] = (_Float16)cosf(xy * m);
          fr[3 + 6 * i + 5] = (_Float16)cosf(xz * m);
        }
#pragma unroll
        for (int j = 57; j < 64; ++j) fr[j] = (_Float16)0.0f;
      }
    }
    asm volatile("" ::: "memory");  // DS ops are in-order per wave; fence the compiler

    // ---- t = relu(freq @ tw0 + tb0): [16x64] -> act cols 160..223 ----
    {
      ASrc as[2] = { {act2, 128}, {act2 + 32, 128} };
      do_layer<2, 4>(as, tw0t, 64, bias + 0, act + 160, 224);
    }
    asm volatile("" ::: "memory");
    // ---- L0: [h|t](96) -> 128, act cols 0..127 ----
    {
      ASrc as[3] = { {act + 128, 224}, {act + 160, 224}, {act + 192, 224} };
      do_layer<3, 8>(as, w0t, 96, bias + 64, act, 224);
    }
    asm volatile("" ::: "memory");
    // ---- L1: 128 -> 128, act2 ----
    {
      ASrc as[4] = { {act, 224}, {act + 32, 224}, {act + 64, 224}, {act + 96, 224} };
      do_layer<4, 8>(as, w1t, 128, bias + 192, act2, 128);
    }
    asm volatile("" ::: "memory");
    // ---- L2: [L1|h|t](224) -> 128, act cols 0..127 (skip concat = extra chunks) ----
    {
      ASrc as[7] = { {act2, 128}, {act2 + 32, 128}, {act2 + 64, 128}, {act2 + 96, 128},
                     {act + 128, 224}, {act + 160, 224}, {act + 192, 224} };
      do_layer<7, 8>(as, w2t, 224, bias + 320, act, 224);
    }
    asm volatile("" ::: "memory");
    // ---- L3: [L2|h|t](224) -> 128, act2 ----
    {
      ASrc as[7] = { {act, 224}, {act + 32, 224}, {act + 64, 224}, {act + 96, 224},
                     {act + 128, 224}, {act + 160, 224}, {act + 192, 224} };
      do_layer<7, 8>(as, w3t, 224, bias + 448, act2, 128);
    }
    asm volatile("" ::: "memory");
    // ---- L4: 128 -> 2 (padded to 16); sigmoid col0, relu col1, store global ----
    {
      const int n  = lane & 15;
      const int mb = hi << 3;
      const float bv = bias[576 + n];
      v8f c;
#pragma unroll
      for (int r = 0; r < 8; ++r) c[r] = bv;
      ASrc as[4] = { {act2, 128}, {act2 + 32, 128}, {act2 + 64, 128}, {act2 + 96, 128} };
#pragma unroll
      for (int kc = 0; kc < 4; ++kc) {
        v16h a = lds_loadA(as[kc].base, as[kc].pitch, lane);
        v16h b = lds_loadB(w4t, 128, 0, kc * 32, lane);
        c = __builtin_amdgcn_wmma_f32_16x16x32_f16(false, a, false, b, (short)0, c,
                                                   false, false);
      }
#pragma unroll
      for (int r = 0; r < 8; ++r) {
        const int gq = pbase + mb + r;
        const float v = c[r];
        if (n == 0)      out[gq]        = 1.0f / (1.0f + __expf(-v));  // colors
        else if (n == 1) out[NPTS + gq] = fmaxf(v, 0.0f);              // densities
      }
    }
  }
}

extern "C" void kernel_launch(void* const* d_in, const int* in_sizes, int n_in,
                              void* d_out, int out_size, void* d_ws, size_t ws_size,
                              hipStream_t stream) {
  const float* x   = (const float*)d_in[0];
  const float* emb = (const float*)d_in[1];
  const float* tw0 = (const float*)d_in[2];
  const float* tb0 = (const float*)d_in[3];
  const float* w0  = (const float*)d_in[4];
  const float* b0  = (const float*)d_in[5];
  const float* w1  = (const float*)d_in[6];
  const float* b1  = (const float*)d_in[7];
  const float* w2  = (const float*)d_in[8];
  const float* b2  = (const float*)d_in[9];
  const float* w3  = (const float*)d_in[10];
  const float* b3  = (const float*)d_in[11];
  const float* w4  = (const float*)d_in[12];
  const float* b4  = (const float*)d_in[13];
  float* out = (float*)d_out;

  (void)hipFuncSetAttribute((const void*)ingp_fused,
                            hipFuncAttributeMaxDynamicSharedMemorySize,
                            (int)SMEM_BYTES);
  ingp_fused<<<dim3(2048), dim3(256), SMEM_BYTES, stream>>>(
      x, emb, tw0, tb0, w0, b0, w1, b1, w2, b2, w3, b3, w4, b4, out);
}
